// MinimalKAN_Normalized_88098369175689
// MI455X (gfx1250) — matrix operand order repaired and checked by reading, not compiled, not run on hardware
//
#include <hip/hip_runtime.h>

// ---------- types ----------
typedef __attribute__((ext_vector_type(16))) _Float16 v16h;
typedef __attribute__((ext_vector_type(8)))  float    v8f;
typedef __attribute__((ext_vector_type(4)))  float    f32x4;
typedef unsigned int u32;
typedef __attribute__((ext_vector_type(4))) u32 u32x4;
typedef __attribute__((ext_vector_type(2))) u32 u32x2;
typedef __attribute__((ext_vector_type(4))) int i32x4;

// address-space qualified pointee types for the async global->LDS builtin
typedef __attribute__((address_space(1))) i32x4 gas_i32x4;   // global
typedef __attribute__((address_space(3))) i32x4 las_i32x4;   // LDS

union Frag { v16h v; u32x4 u[2]; };

// ---------- problem sizes ----------
#define B_SZ 32768
#define I_SZ 512
#define O_SZ 512
#define K_SZ 1536          // 3 * I  (x, x^2, x^3 slabs)
#define BM   128           // M tile per workgroup
#define BN   128           // N tile per workgroup
#define BK   32            // K step (one f16 WMMA depth)
#define LDP  40            // padded LDS row stride in halves (80 B)
#define NSTEP (K_SZ / BK)  // 48 K-steps

// ---------- CDNA5 async global->LDS path (probe via __has_builtin) ----------
#if defined(__has_builtin)
#  if __has_builtin(__builtin_amdgcn_global_load_async_to_lds_b128)
#    define HAS_ASYNC_LDS 1
#  endif
#endif
#ifndef HAS_ASYNC_LDS
#  define HAS_ASYNC_LDS 0
#endif

__device__ __forceinline__ void async_b128(const _Float16* gsrc, _Float16* ldst) {
#if HAS_ASYNC_LDS
    // GLOBAL_LOAD_ASYNC_TO_LDS_B128: 16 B per lane, tracked by ASYNCcnt
    __builtin_amdgcn_global_load_async_to_lds_b128(
        (gas_i32x4*)gsrc,
        (las_i32x4*)ldst,
        0 /*imm offset*/, 0 /*cpol*/);
#else
    *(u32x4*)ldst = *(const u32x4*)gsrc;
#endif
}

__device__ __forceinline__ void wait_async_all() {
#if HAS_ASYNC_LDS
#  if __has_builtin(__builtin_amdgcn_s_wait_asynccnt)
    __builtin_amdgcn_s_wait_asynccnt(0);
#  else
    asm volatile("s_wait_asynccnt 0x0" ::: "memory");
#  endif
#endif
}

__device__ __forceinline__ float sigmoidf(float v) {
    return 1.0f / (1.0f + __expf(-v));
}

// ---------------------------------------------------------------------------
// Kernel 1: build combined f16 weight matrix Wt[o][k], k-contiguous, (O, 3I).
//   k in [0,512)    : (1-a)*W[o][i] + (a/sqrt(I))*coeffs[i][o][0]   (times x)
//   k in [512,1024) :                 (a/sqrt(I))*coeffs[i][o][1]   (times x^2)
//   k in [1024,1536):                 (a/sqrt(I))*coeffs[i][o][2]   (times x^3)
// ---------------------------------------------------------------------------
__global__ void kan_prep_kernel(const float* __restrict__ W,
                                const float* __restrict__ coeffs,
                                const float* __restrict__ alpha,
                                _Float16* __restrict__ Wt) {
    int idx = blockIdx.x * blockDim.x + threadIdx.x;   // 0 .. O*I-1
    int o = idx >> 9;
    int i = idx & 511;
    float a  = sigmoidf(alpha[0]);
    float s0 = 1.0f - a;
    float sk = a * 0.04419417382415922f;               // 1/sqrt(512)
    float w  = W[o * I_SZ + i];
    const float* cf = coeffs + ((size_t)i * O_SZ + o) * 3;
    _Float16* row = Wt + (size_t)o * K_SZ;
    row[i]        = (_Float16)(s0 * w + sk * cf[0]);
    row[i + 512]  = (_Float16)(sk * cf[1]);
    row[i + 1024] = (_Float16)(sk * cf[2]);
}

// ---------------------------------------------------------------------------
// Kernel 2: fused KAN GEMM, double-buffered LDS + async B-tile copies.
//   out[b][o] = sum_k A[b][k] * Wt[o][k] + (1-a)*bias[o],  A[b][k] = x^p
// 128x128 tile / 256 threads (8 waves: 4 along M x 2 along N),
// each wave: 2x4 accumulators of v_wmma_f32_16x16x32_f16.
// ---------------------------------------------------------------------------
__global__ void __launch_bounds__(256)
kan_gemm_kernel(const float* __restrict__ x,
                const _Float16* __restrict__ Wt,
                const float* __restrict__ bias,
                const float* __restrict__ alpha,
                float* __restrict__ out) {
    __shared__ _Float16 As[2][BM * LDP];   // A tiles, row-major [m][k], padded
    __shared__ _Float16 Bs[2][BN * LDP];   // B tiles, [n][k] (k contiguous), padded

    const int tid   = threadIdx.x;
    const int lane  = tid & 31;
    const int wave  = tid >> 5;
    const int waveM = wave & 3;          // 4 waves along M (32 rows each)
    const int waveN = wave >> 2;         // 2 waves along N (64 cols each)
    const int l15   = lane & 15;
    const int lhi   = lane >> 4;         // 0 or 1

    const int rowBase = blockIdx.y * BM;
    const int colBase = blockIdx.x * BN;

    v8f acc[2][4];
    const v8f vzero = {0.f, 0.f, 0.f, 0.f, 0.f, 0.f, 0.f, 0.f};
#pragma unroll
    for (int mt = 0; mt < 2; ++mt)
#pragma unroll
        for (int nt = 0; nt < 4; ++nt)
            acc[mt][nt] = vzero;

    // global->LDS loader coordinates
    const int arow = tid >> 3;           // 0..31 (row within a 32-row pass)
    const int acol = (tid & 7) * 4;      // 0,4,..,28 (f32 quad within K chunk)

    // ---- A tile: 128 x 32, read x, raise to slab power, convert to f16 ----
    auto loadA = [&](int c, int buf) {
        const int p  = c >> 4;                   // 0,1,2 -> power 1,2,3
        const int i0 = (c & 15) * BK;            // column window inside x
#pragma unroll
        for (int pass = 0; pass < 4; ++pass) {
            const int r = pass * 32 + arow;
            f32x4 v = *(const f32x4*)(x + (size_t)(rowBase + r) * I_SZ + i0 + acol);
            f32x4 w = v;
            if (p > 0) w = w * v;                // x^2
            if (p > 1) w = w * v;                // x^3
            _Float16 h4[4];
            h4[0] = (_Float16)w.x; h4[1] = (_Float16)w.y;
            h4[2] = (_Float16)w.z; h4[3] = (_Float16)w.w;
            *(u32x2*)(&As[buf][r * LDP + acol]) = *(const u32x2*)h4;
        }
    };

    // ---- B tile: 128 x 32 f16, pure copy -> async global->LDS (2 x b128/thread)
    auto loadB = [&](int c, int buf) {
#pragma unroll
        for (int j = 0; j < 2; ++j) {
            const int id = tid * 2 + j;          // 0..511 16-byte transfers
            const int n  = id >> 2;              // 0..127
            const int kq = (id & 3) * 8;         // 0,8,16,24 halves
            async_b128(Wt + (size_t)(colBase + n) * K_SZ + c * BK + kq,
                       &Bs[buf][n * LDP + kq]);
        }
    };

    // ---- software pipeline: prologue ----
    loadB(0, 0);
    loadA(0, 0);

    for (int c = 0; c < NSTEP; ++c) {
        const int cur = c & 1;

        wait_async_all();      // our async B copies into buf 'cur' are done
        __syncthreads();       // tile 'cur' published; buf 'cur^1' free

        if (c + 1 < NSTEP) {   // start next tile's traffic before computing
            loadB(c + 1, cur ^ 1);
            loadA(c + 1, cur ^ 1);
        }

        // ---- build WMMA fragments from LDS (two b128 loads each) ----
        // A 16-bit 16x32 layout: lanes 0-15 hold K 0-7/16-23, lanes 16-31 hold K 8-15/24-31
        Frag af[2];
#pragma unroll
        for (int mt = 0; mt < 2; ++mt) {
            const int m  = waveM * 32 + mt * 16 + l15;
            const int ka = lhi * 8;
            af[mt].u[0] = *(const u32x4*)(&As[cur][m * LDP + ka]);
            af[mt].u[1] = *(const u32x4*)(&As[cur][m * LDP + 16 + ka]);
        }
        // B 16-bit 32x16 layout: lanes 0-15 hold K 0-7/16-23, lanes 16-31 hold K 8-15/24-31
        Frag bf[4];
#pragma unroll
        for (int nt = 0; nt < 4; ++nt) {
            const int n  = waveN * 64 + nt * 16 + l15;
            const int kb = lhi * 8;
            bf[nt].u[0] = *(const u32x4*)(&Bs[cur][n * LDP + kb]);
            bf[nt].u[1] = *(const u32x4*)(&Bs[cur][n * LDP + kb + 16]);
        }

        // ---- 8 WMMAs per wave per K step ----
#pragma unroll
        for (int mt = 0; mt < 2; ++mt)
#pragma unroll
            for (int nt = 0; nt < 4; ++nt)
                acc[mt][nt] = __builtin_amdgcn_wmma_f32_16x16x32_f16(
                    false, af[mt].v, false, bf[nt].v,
                    (short)0, acc[mt][nt], false, false);
    }

    // ---- epilogue: add (1-a)*bias, write f32 results ----
    // C/D 16x16 f32 layout: VGPR r, lanes 0-15 -> M=r, lanes 16-31 -> M=8+r, N=lane&15
    const float a  = sigmoidf(alpha[0]);
    const float s0 = 1.0f - a;
#pragma unroll
    for (int nt = 0; nt < 4; ++nt) {
        const int col = colBase + waveN * 64 + nt * 16 + l15;
        const float bv = s0 * bias[col];
#pragma unroll
        for (int mt = 0; mt < 2; ++mt) {
            const int row0 = rowBase + waveM * 32 + mt * 16 + lhi * 8;
#pragma unroll
            for (int r = 0; r < 8; ++r) {
                out[(size_t)(row0 + r) * O_SZ + col] = acc[mt][nt][r] + bv;
            }
        }
    }
}

// ---------------------------------------------------------------------------
extern "C" void kernel_launch(void* const* d_in, const int* in_sizes, int n_in,
                              void* d_out, int out_size, void* d_ws, size_t ws_size,
                              hipStream_t stream) {
    (void)in_sizes; (void)n_in; (void)out_size; (void)ws_size;
    const float* x      = (const float*)d_in[0];   // (B, I) f32
    const float* coeffs = (const float*)d_in[1];   // (I, O, 3) f32
    const float* W      = (const float*)d_in[2];   // (O, I) f32
    const float* bias   = (const float*)d_in[3];   // (O,) f32
    const float* alpha  = (const float*)d_in[4];   // scalar f32
    float* out = (float*)d_out;                    // (B, O) f32
    _Float16* Wt = (_Float16*)d_ws;                // (O, 3I) f16 = 1.5 MB scratch

    kan_prep_kernel<<<dim3((O_SZ * I_SZ) / 256), dim3(256), 0, stream>>>(
        W, coeffs, alpha, Wt);

    kan_gemm_kernel<<<dim3(O_SZ / BN, B_SZ / BM), dim3(256), 0, stream>>>(
        x, Wt, bias, alpha, out);
}
